// MultiHeadAttention_15676630630660
// MI455X (gfx1250) — matrix operand order repaired and checked by reading, not compiled, not run on hardware
//
#include <hip/hip_runtime.h>
#include <hip/hip_bf16.h>

// MHA block for MI455X (gfx1250): bf16 WMMA GEMMs + flash-style attention with
// double-buffered GLOBAL_LOAD_ASYNC_TO_LDS staging of K/V tiles (ASYNCcnt).
#define SEQ   2048
#define DM    512
#define DK    64
#define NH    8
#define BATCH 4

typedef __bf16 bf16;
typedef __attribute__((ext_vector_type(16))) __bf16 v16bf;
typedef __attribute__((ext_vector_type(8)))  __bf16 v8bf;
typedef __attribute__((ext_vector_type(8)))  float  v8f;

__device__ __forceinline__ v8f wmma_bf16(v16bf a, v16bf b, v8f c) {
  return __builtin_amdgcn_wmma_f32_16x16x32_bf16(false, a, false, b, (short)0, c,
                                                 false, false);
}

// Async DMA: 16 bytes global -> LDS, tracked with ASYNCcnt (per-wave, in-order).
__device__ __forceinline__ void async_copy_b128(uint32_t lds_off, const bf16* g) {
  asm volatile("global_load_async_to_lds_b128 %0, %1, off"
               :: "v"(lds_off), "v"(g) : "memory");
}

// A-fragment (16x32, M = lane%16): elems 0..7 = cols k0+half*8.., 8..15 = cols k0+16+half*8..
__device__ __forceinline__ v16bf frag_a_f32(const float* p, int half) {
  const float4* lo = (const float4*)(p + half * 8);
  const float4* hi = (const float4*)(p + 16 + half * 8);
  float4 a0 = lo[0], a1 = lo[1], b0 = hi[0], b1 = hi[1];
  v16bf r;
  r[0]=(bf16)a0.x; r[1]=(bf16)a0.y; r[2]=(bf16)a0.z; r[3]=(bf16)a0.w;
  r[4]=(bf16)a1.x; r[5]=(bf16)a1.y; r[6]=(bf16)a1.z; r[7]=(bf16)a1.w;
  r[8]=(bf16)b0.x; r[9]=(bf16)b0.y; r[10]=(bf16)b0.z; r[11]=(bf16)b0.w;
  r[12]=(bf16)b1.x; r[13]=(bf16)b1.y; r[14]=(bf16)b1.z; r[15]=(bf16)b1.w;
  return r;
}

__device__ __forceinline__ v16bf cat8(v8bf lo, v8bf hi) {
  v16bf r;
#pragma unroll
  for (int i = 0; i < 8; ++i) { r[i] = lo[i]; r[8 + i] = hi[i]; }
  return r;
}

__device__ __forceinline__ v16bf frag_a_bf16(const bf16* p, int half) {
  v8bf lo = *(const v8bf*)(p + half * 8);
  v8bf hi = *(const v8bf*)(p + 16 + half * 8);
  return cat8(lo, hi);
}

// B-fragment (32x16, N = lane%16): elems 0..7 = rows K = half*8.., 8..15 = rows K = 16+half*8..
__device__ __forceinline__ v16bf frag_b_f32(const float* p, int half) {
  const float4* q = (const float4*)(p + half * 8);
  const float4* q2 = (const float4*)(p + 16 + half * 8);
  float4 a0 = q[0], a1 = q[1], a2 = q2[0], a3 = q2[1];
  v16bf r;
  r[0]=(bf16)a0.x; r[1]=(bf16)a0.y; r[2]=(bf16)a0.z; r[3]=(bf16)a0.w;
  r[4]=(bf16)a1.x; r[5]=(bf16)a1.y; r[6]=(bf16)a1.z; r[7]=(bf16)a1.w;
  r[8]=(bf16)a2.x; r[9]=(bf16)a2.y; r[10]=(bf16)a2.z; r[11]=(bf16)a2.w;
  r[12]=(bf16)a3.x; r[13]=(bf16)a3.y; r[14]=(bf16)a3.z; r[15]=(bf16)a3.w;
  return r;
}

__device__ __forceinline__ v16bf frag_b_bf16(const bf16* p, int half) {
  v8bf lo = *(const v8bf*)(p + half * 8);
  v8bf hi = *(const v8bf*)(p + 16 + half * 8);
  return cat8(lo, hi);
}

// ---------------- kernel 1: Q/K/V projections (x @ W.T + b) -> bf16 ----------------
// q,k: [B,H,S,DK]; v stored transposed: [B,H,DK,S] (so attn context B-frags are contiguous)
__global__ void proj_kernel(const float* __restrict__ Xq, const float* __restrict__ Xk,
                            const float* __restrict__ Xv, const float* __restrict__ Wq,
                            const float* __restrict__ Wk, const float* __restrict__ Wv,
                            const float* __restrict__ bq, const float* __restrict__ bk,
                            const float* __restrict__ bv, bf16* __restrict__ qb,
                            bf16* __restrict__ kb, bf16* __restrict__ vt) {
  const int wave = threadIdx.x >> 5;
  const int lane = threadIdx.x & 31;
  const int idx = lane & 15;
  const int half = lane >> 4;
  const float *X, *W, *bias;
  bf16* dst;
  int mode;
  if (blockIdx.y == 0)      { X = Xq; W = Wq; bias = bq; dst = qb; mode = 0; }
  else if (blockIdx.y == 1) { X = Xk; W = Wk; bias = bk; dst = kb; mode = 0; }
  else                      { X = Xv; W = Wv; bias = bv; dst = vt; mode = 1; }
  const int tile = blockIdx.x * 8 + wave;
  const int m0 = (tile >> 5) << 4;  // 512 M-tiles
  const int n0 = (tile & 31) << 4;  // 32 N-tiles
  const float* arow = X + (size_t)(m0 + idx) * DM;
  const float* brow = W + (size_t)(n0 + idx) * DM;
  v8f c = {};
#pragma unroll 4
  for (int k0 = 0; k0 < DM; k0 += 32) {
    c = wmma_bf16(frag_a_f32(arow + k0, half), frag_b_f32(brow + k0, half), c);
  }
  const int col = n0 + idx;
  const float bias_v = bias[col];
  const int h = col >> 6, dk = col & 63;
#pragma unroll
  for (int r = 0; r < 8; ++r) {
    const int m = m0 + r + 8 * half;
    const int bb = m >> 11, s = m & 2047;
    const bf16 v = (bf16)(c[r] + bias_v);
    if (mode == 0)
      dst[(((size_t)bb * NH + h) * SEQ + s) * DK + dk] = v;
    else
      dst[(((size_t)bb * NH + h) * DK + dk) * SEQ + s] = v;
  }
}

// ---------------- kernel 2: flash attention ----------------
// Block = 8 waves, one (b,h), 128 q-rows. K/V tiles (32 keys) staged cooperatively
// in LDS via double-buffered async-to-LDS DMA; scores/context on WMMA.
__global__ void attn_kernel(const bf16* __restrict__ qb, const bf16* __restrict__ kb,
                            const bf16* __restrict__ vt, bf16* __restrict__ ctx) {
  __shared__ __align__(32) bf16 kstage[2][32 * 64];  // 2 x 4KB: keys x d
  __shared__ __align__(32) bf16 vstage[2][64 * 32];  // 2 x 4KB: d x keys
  __shared__ __align__(32) bf16 pstage[8][16 * 32];  // per-wave P (C-layout -> A-frag)
  const int tid = threadIdx.x;
  const int wave = tid >> 5, lane = tid & 31, idx = lane & 15, half = lane >> 4;
  const int bh = blockIdx.y;  // 0..31 = b*8+h
  const int b = bh >> 3, h = bh & 7;
  const int m0 = blockIdx.x * 128 + wave * 16;

  const bf16* kg = kb + (size_t)bh * SEQ * DK;  // [2048][64], tiles are contiguous 4KB
  const bf16* vg = vt + (size_t)bh * DK * SEQ;  // [64][2048]

  // per-thread async transfer coordinates (2 x b128 per thread per tile)
  const uint32_t klds0 = (uint32_t)(uintptr_t)(&kstage[0][0]) + (uint32_t)tid * 16u;
  const uint32_t klds1 = (uint32_t)(uintptr_t)(&kstage[1][0]) + (uint32_t)tid * 16u;
  const int vrow = tid >> 2, vch = tid & 3;
  const uint32_t voff = (uint32_t)vrow * 64u + (uint32_t)vch * 16u;
  const uint32_t vlds0 = (uint32_t)(uintptr_t)(&vstage[0][0]) + voff;
  const uint32_t vlds1 = (uint32_t)(uintptr_t)(&vstage[1][0]) + voff;
  const bf16* vgrow = vg + (size_t)vrow * SEQ + vch * 8;

  const bf16* qbase = qb + ((size_t)bh * SEQ + m0 + idx) * DK;
  const v16bf aq0 = frag_a_bf16(qbase, half);       // d = 0..31
  const v16bf aq1 = frag_a_bf16(qbase + 32, half);  // d = 32..63

  float mrow[8], lrow[8];
#pragma unroll
  for (int r = 0; r < 8; ++r) { mrow[r] = -1e30f; lrow[r] = 0.f; }
  v8f o0 = {}, o1 = {}, o2 = {}, o3 = {};
  bf16* pst = &pstage[wave][0];

  // prologue: tile 0 -> buffer 0
  async_copy_b128(klds0, kg + tid * 8);
  async_copy_b128(vlds0, vgrow);

  for (int it = 0; it < SEQ / 32; ++it) {
    const int k0 = it * 32;
    const int cur = it & 1;
    if (it + 1 < SEQ / 32) {
      // prefetch next tile into the other buffer, then retire only the current
      // tile's 2 ops (per-wave in-order completion => ASYNCcnt <= 2 suffices)
      const int nk0 = k0 + 32;
      async_copy_b128(cur ? klds0 : klds1, kg + (size_t)nk0 * DK + tid * 8);
      async_copy_b128(cur ? vlds0 : vlds1, vgrow + nk0);
      asm volatile("s_wait_asynccnt 0x2" ::: "memory");
    } else {
      asm volatile("s_wait_asynccnt 0x0" ::: "memory");
    }
    __syncthreads();  // current tile visible to all waves

    const bf16* kst = &kstage[cur][0];
    const bf16* vst = &vstage[cur][0];

    // S tile: 16 queries x 32 keys (two 16-key column tiles), K-dim = 64 over d
    v8f s0 = {}, s1 = {};
    s0 = wmma_bf16(aq0, frag_b_bf16(kst + (idx) * 64, half), s0);
    s0 = wmma_bf16(aq1, frag_b_bf16(kst + (idx) * 64 + 32, half), s0);
    s1 = wmma_bf16(aq0, frag_b_bf16(kst + (16 + idx) * 64, half), s1);
    s1 = wmma_bf16(aq1, frag_b_bf16(kst + (16 + idx) * 64 + 32, half), s1);

    // online softmax: row r lives in this lane's 16-lane half -> shfl width 16
#pragma unroll
    for (int r = 0; r < 8; ++r) {
      const float x0 = s0[r] * 0.125f, x1 = s1[r] * 0.125f;  // 1/sqrt(64)
      float vmax = fmaxf(x0, x1);
#pragma unroll
      for (int off = 1; off < 16; off <<= 1) vmax = fmaxf(vmax, __shfl_xor(vmax, off, 16));
      const float mnew = fmaxf(mrow[r], vmax);
      const float alpha = __expf(mrow[r] - mnew);
      mrow[r] = mnew;
      const float p0 = __expf(x0 - mnew), p1 = __expf(x1 - mnew);
      float rs = p0 + p1;
#pragma unroll
      for (int off = 1; off < 16; off <<= 1) rs += __shfl_xor(rs, off, 16);
      lrow[r] = lrow[r] * alpha + rs;
      o0[r] *= alpha; o1[r] *= alpha; o2[r] *= alpha; o3[r] *= alpha;
      const int row = r + 8 * half;
      pst[row * 32 + idx] = (bf16)p0;
      pst[row * 32 + 16 + idx] = (bf16)p1;
    }
    // wave-local LDS RAW: per-wave staging + in-order LDS => dscnt wait suffices
    asm volatile("s_wait_dscnt 0x0" ::: "memory");
    const v16bf ap = frag_a_bf16(pst + idx * 32, half);  // P as A-frag (16x32)
    o0 = wmma_bf16(ap, frag_b_bf16(vst + (idx) * 32, half), o0);
    o1 = wmma_bf16(ap, frag_b_bf16(vst + (16 + idx) * 32, half), o1);
    o2 = wmma_bf16(ap, frag_b_bf16(vst + (32 + idx) * 32, half), o2);
    o3 = wmma_bf16(ap, frag_b_bf16(vst + (48 + idx) * 32, half), o3);

    __syncthreads();  // all waves done reading before this buffer is refilled
  }

#pragma unroll
  for (int r = 0; r < 8; ++r) {
    const float inv = 1.0f / lrow[r];
    const int row = m0 + r + 8 * half;
    const size_t base = ((size_t)b * SEQ + row) * DM + h * DK + idx;
    ctx[base + 0] = (bf16)(o0[r] * inv);
    ctx[base + 16] = (bf16)(o1[r] * inv);
    ctx[base + 32] = (bf16)(o2[r] * inv);
    ctx[base + 48] = (bf16)(o3[r] * inv);
  }
}

// ---------------- kernel 3: ctx @ Wo.T + bo + residual, then LayerNorm ----------------
__global__ void outln_kernel(const bf16* __restrict__ ctx, const float* __restrict__ Wo,
                             const float* __restrict__ bo, const float* __restrict__ Qin,
                             const float* __restrict__ gamma, const float* __restrict__ beta,
                             float* __restrict__ out) {
  __shared__ float xrow[16][DM];  // 32 KB
  const int wave = threadIdx.x >> 5;
  const int lane = threadIdx.x & 31;
  const int idx = lane & 15;
  const int half = lane >> 4;
  const int m0 = blockIdx.x * 16;
  const bf16* arow = ctx + (size_t)(m0 + idx) * DM;
  const int nb = wave * 64;  // 8 waves x 64 cols = 512
  const float* w0 = Wo + (size_t)(nb + 0 + idx) * DM;
  const float* w1 = Wo + (size_t)(nb + 16 + idx) * DM;
  const float* w2 = Wo + (size_t)(nb + 32 + idx) * DM;
  const float* w3 = Wo + (size_t)(nb + 48 + idx) * DM;
  v8f c0 = {}, c1 = {}, c2 = {}, c3 = {};
  for (int k0 = 0; k0 < DM; k0 += 32) {
    const v16bf a = frag_a_bf16(arow + k0, half);
    c0 = wmma_bf16(a, frag_b_f32(w0 + k0, half), c0);
    c1 = wmma_bf16(a, frag_b_f32(w1 + k0, half), c1);
    c2 = wmma_bf16(a, frag_b_f32(w2 + k0, half), c2);
    c3 = wmma_bf16(a, frag_b_f32(w3 + k0, half), c3);
  }
#pragma unroll
  for (int t = 0; t < 4; ++t) {
    const v8f c = t == 0 ? c0 : t == 1 ? c1 : t == 2 ? c2 : c3;
    const int col = nb + t * 16 + idx;
    const float bias_v = bo[col];
#pragma unroll
    for (int r = 0; r < 8; ++r) {
      const int row = r + 8 * half;
      xrow[row][col] = c[r] + bias_v + Qin[(size_t)(m0 + row) * DM + col];
    }
  }
  __syncthreads();
  const int row = threadIdx.x >> 4;  // 16 rows, 16 threads each
  const int cl = threadIdx.x & 15;
  float sum = 0.f, sq = 0.f;
#pragma unroll
  for (int j = 0; j < 32; ++j) {
    const float x = xrow[row][cl + j * 16];
    sum += x; sq += x * x;
  }
#pragma unroll
  for (int off = 1; off < 16; off <<= 1) {
    sum += __shfl_xor(sum, off, 16);
    sq += __shfl_xor(sq, off, 16);
  }
  const float mean = sum * (1.0f / DM);
  const float var = sq * (1.0f / DM) - mean * mean;
  const float rs = rsqrtf(var + 1e-5f);
#pragma unroll
  for (int j = 0; j < 32; ++j) {
    const int col = cl + j * 16;
    const float x = xrow[row][col];
    out[(size_t)(m0 + row) * DM + col] = (x - mean) * rs * gamma[col] + beta[col];
  }
}

extern "C" void kernel_launch(void* const* d_in, const int* in_sizes, int n_in,
                              void* d_out, int out_size, void* d_ws, size_t ws_size,
                              hipStream_t stream) {
  const float* Q  = (const float*)d_in[0];
  const float* K  = (const float*)d_in[1];
  const float* V  = (const float*)d_in[2];
  const float* Wq = (const float*)d_in[3];
  const float* bq = (const float*)d_in[4];
  const float* Wk = (const float*)d_in[5];
  const float* bk = (const float*)d_in[6];
  const float* Wv = (const float*)d_in[7];
  const float* bv = (const float*)d_in[8];
  const float* Wo = (const float*)d_in[9];
  const float* bo = (const float*)d_in[10];
  const float* gamma = (const float*)d_in[11];
  const float* beta  = (const float*)d_in[12];
  float* out = (float*)d_out;

  char* ws = (char*)d_ws;
  const size_t NE = (size_t)BATCH * SEQ * DM;  // 4,194,304 elems
  bf16* qb  = (bf16*)(ws + 0 * NE);            // 8 MB each (bf16)
  bf16* kb  = (bf16*)(ws + 2 * NE);
  bf16* vt  = (bf16*)(ws + 4 * NE);
  bf16* ctx = (bf16*)(ws + 6 * NE);

  // 1) Q/K/V projections: 3 x 16384 (16x16) tiles, 8 waves/block
  proj_kernel<<<dim3(2048, 3), 256, 0, stream>>>(Q, K, V, Wq, Wk, Wv, bq, bk, bv,
                                                 qb, kb, vt);
  // 2) flash attention: grid = (S/128, B*H), 8 waves/block, 16 q-rows/wave
  attn_kernel<<<dim3(16, 32), 256, 0, stream>>>(qb, kb, vt, ctx);
  // 3) output projection + residual + layernorm: 16 rows/block
  outln_kernel<<<dim3(512), 256, 0, stream>>>(ctx, Wo, bo, Q, gamma, beta, out);
}
